// SAE_22591527977427
// MI455X (gfx1250) — matrix operand order repaired and checked by reading, not compiled, not run on hardware
//
#include <hip/hip_runtime.h>
#include <hip/hip_bf16.h>
#include <math.h>

// ---------------------------------------------------------------------------
// TopK-SAE forward for MI455X (gfx1250, wave32, WMMA + Tensor Data Mover).
//   split : x-bd and Ae -> bf16 hi/lo arrays (one-time, ~190 MB traffic)
//   gemm  : pre = relu((x-bd) @ Ae^T), split-bf16 3-product WMMA,
//           tiles DMA'd into LDS by TDM (tensor_load_to_lds, TENSORcnt),
//           double-buffered, inner loop = ds_load fragments + v_wmma only
//   topk  : per-row top-64 via 4-level radix threshold histograms
//   decode: out = bd + lam * sum v_e/(eps+||Ae_j||) * Ae_j   (Ad == Ae^T)
//
// Workspace layout (requires ws_size >= ~911 MB):
//   pre   : 8192*24576 f32      (805,306,368 B)
//   norms : 24576 f32
//   topv  : 8192*64 f32
//   topi  : 8192*64 i32
//   xh,xl : 8192*768 bf16 each
//   Aeh,Ael: 24576*768 bf16 each
// ---------------------------------------------------------------------------

#define BATCH 8192
#define WIDTH 24576
#define DIMIN 768
#define KSEL  64

#define BM 128
#define BN 128
#define BK 32
#define NKSTEP (DIMIN / BK)   // 24
#define LDSTRIDE 40           // 32 bf16 + 8 pad (80 B row pitch, matches TDM pad)

typedef __attribute__((ext_vector_type(16))) __bf16 v16bf;
typedef __attribute__((ext_vector_type(8)))  float  v8f;
typedef __attribute__((ext_vector_type(4)))  unsigned int u32x4;
typedef __attribute__((ext_vector_type(8)))  int i32x8;
typedef __attribute__((ext_vector_type(4)))  int i32x4;

__device__ __forceinline__ unsigned short bf16_rn_bits(float f) {
    unsigned u = __builtin_bit_cast(unsigned, f);
    unsigned r = u + 0x7FFFu + ((u >> 16) & 1u);
    return (unsigned short)(r >> 16);
}
__device__ __forceinline__ float bf16_bits_to_f32(unsigned short h) {
    return __builtin_bit_cast(float, (unsigned)h << 16);
}
__device__ __forceinline__ void split_bf16(float f, unsigned short& h, unsigned short& l) {
    h = bf16_rn_bits(f);
    float lo = f - bf16_bits_to_f32(h);
    l = bf16_rn_bits(lo);
}

// ---------------------------------------------------------------------------
// Kernel 0: one-time split pass. Rows [0,BATCH) = x - bd, rows [BATCH, ..) = Ae.
// ---------------------------------------------------------------------------
__global__ __launch_bounds__(256) void sae_split_kernel(
    const float* __restrict__ x, const float* __restrict__ Ae,
    const float* __restrict__ bd,
    unsigned short* __restrict__ xh, unsigned short* __restrict__ xl,
    unsigned short* __restrict__ Aeh, unsigned short* __restrict__ Ael) {
    const int row = blockIdx.x;
    const int tid = threadIdx.x;
    if (row < BATCH) {
        const size_t base = (size_t)row * DIMIN;
        for (int c = tid; c < DIMIN; c += 256) {
            unsigned short h, l;
            split_bf16(x[base + c] - bd[c], h, l);
            xh[base + c] = h; xl[base + c] = l;
        }
    } else {
        const size_t base = (size_t)(row - BATCH) * DIMIN;
        for (int c = tid; c < DIMIN; c += 256) {
            unsigned short h, l;
            split_bf16(Ae[base + c], h, l);
            Aeh[base + c] = h; Ael[base + c] = l;
        }
    }
}

// ---------------------------------------------------------------------------
// Kernel 1: decoder column norms = row norms of Ae.
// ---------------------------------------------------------------------------
__global__ __launch_bounds__(256) void sae_norms_kernel(
    const float* __restrict__ Ae, float* __restrict__ norms) {
    const int row = blockIdx.x;
    const int tid = threadIdx.x;
    __shared__ float red[256];
    const float* a = Ae + (size_t)row * DIMIN;
    float v0 = a[tid];
    float v1 = a[tid + 256];
    float v2 = a[tid + 512];
    red[tid] = v0 * v0 + v1 * v1 + v2 * v2;
    __syncthreads();
    for (int o = 128; o > 0; o >>= 1) {
        if (tid < o) red[tid] += red[tid + o];
        __syncthreads();
    }
    if (tid == 0) norms[row] = sqrtf(red[0]);
}

// ---------------------------------------------------------------------------
// TDM: issue one 2-D tile load (128 rows x 32 bf16, row stride 768 bf16) from
// global bf16 tensor into LDS with hardware padding: after every 16 dwords
// (64 B = one tile row) insert 4 dwords (16 B) -> 80 B LDS row pitch.
// Descriptor packing per CDNA5 D# (group0 128b, group1 256b); groups 2/3 and
// the trailing descriptor words are zero (2-D tile).
// ---------------------------------------------------------------------------
__device__ __forceinline__ void tdm_load_tile(const unsigned short* gptr,
                                              unsigned ldsOff) {
    unsigned long long ga = (unsigned long long)(uintptr_t)gptr;
    u32x4 g0;
    g0[0] = 1u;                                   // count=1, user descriptor
    g0[1] = ldsOff;                               // lds_addr (bytes)
    g0[2] = (unsigned)ga;                         // global_addr[31:0]
    g0[3] = (unsigned)((ga >> 32) & 0x1FFFFFFu)   // global_addr[56:32]
            | (2u << 30);                         // type=2 ("image")
    i32x8 g1;
    g1[0] = (int)((1u << 16)                      // data_size=1 -> 2 bytes
                  | (1u << 20)                    // pad_enable
                  | (3u << 22)                    // pad_interval: 16 dwords
                  | (3u << 25));                  // pad_amount: 4 dwords
    g1[1] = (int)(768u << 16);                    // tensor_dim0 = 768 (lo16)
    g1[2] = (int)(128u << 16);                    // dim0 hi=0 | tensor_dim1=128
    g1[3] = (int)(32u << 16);                     // dim1 hi=0 | tile_dim0=32
    g1[4] = 128;                                  // tile_dim1=128, tile_dim2=0
    g1[5] = 768;                                  // tensor_dim0_stride lo32
    g1[6] = 0;
    g1[7] = 0;
    i32x4 z4 = {0, 0, 0, 0};
    i32x8 z8 = {0, 0, 0, 0, 0, 0, 0, 0};
    __builtin_amdgcn_tensor_load_to_lds(g0, g1, z4, z4, z8, 0);
}

// ---------------------------------------------------------------------------
// Kernel 2: encode GEMM, TDM double-buffered, split-bf16 WMMA.
// Grid (WIDTH/BN, BATCH/BM), 256 threads = 8 waves (2 M x 4 N), each wave
// owns a 64x32 subtile = 4x2 WMMA tiles; 24 v_wmma per wave per K step.
// ---------------------------------------------------------------------------
__global__ __launch_bounds__(256) void sae_encode_gemm(
    const unsigned short* __restrict__ xh, const unsigned short* __restrict__ xl,
    const unsigned short* __restrict__ Aeh, const unsigned short* __restrict__ Ael,
    float* __restrict__ pre) {

    // [buf][tensor: 0=Ah 1=Al 2=Bh 3=Bl][128 rows * 40 bf16]
    __shared__ __bf16 smem[2][4][BM * LDSTRIDE];

    const int tid     = threadIdx.x;
    const int lane    = tid & 31;
    const int wave    = tid >> 5;
    const int waveM   = wave >> 2;
    const int waveN   = wave & 3;
    const int lh      = lane >> 4;
    const int l15     = lane & 15;
    const int rowBase = blockIdx.y * BM;
    const int colBase = blockIdx.x * BN;

    v8f acc[4][2];
    for (int mt = 0; mt < 4; ++mt)
        for (int nt = 0; nt < 2; ++nt)
            for (int e = 0; e < 8; ++e) acc[mt][nt][e] = 0.0f;

    const size_t aOff = (size_t)rowBase * DIMIN;
    const size_t bOff = (size_t)colBase * DIMIN;

    // Prologue: wave 0 kicks off the k=0 tiles.
    if (tid < 32) {
        tdm_load_tile(xh  + aOff, (unsigned)(uintptr_t)&smem[0][0][0]);
        tdm_load_tile(xl  + aOff, (unsigned)(uintptr_t)&smem[0][1][0]);
        tdm_load_tile(Aeh + bOff, (unsigned)(uintptr_t)&smem[0][2][0]);
        tdm_load_tile(Ael + bOff, (unsigned)(uintptr_t)&smem[0][3][0]);
    }

    for (int kt = 0; kt < NKSTEP; ++kt) {
        const int cur = kt & 1;
        const int nxt = cur ^ 1;

        __syncthreads();  // (a) everyone done reading smem[nxt] (iter kt-1)

        if (tid < 32) {
            if (kt + 1 < NKSTEP) {
                const size_t k1 = (size_t)(kt + 1) * BK;
                tdm_load_tile(xh  + aOff + k1, (unsigned)(uintptr_t)&smem[nxt][0][0]);
                tdm_load_tile(xl  + aOff + k1, (unsigned)(uintptr_t)&smem[nxt][1][0]);
                tdm_load_tile(Aeh + bOff + k1, (unsigned)(uintptr_t)&smem[nxt][2][0]);
                tdm_load_tile(Ael + bOff + k1, (unsigned)(uintptr_t)&smem[nxt][3][0]);
                __builtin_amdgcn_s_wait_tensorcnt(4);  // cur's 4 ops retired
            } else {
                __builtin_amdgcn_s_wait_tensorcnt(0);
            }
        }
        __syncthreads();  // (b) smem[cur] visible to all waves

        // ---- fragments from smem[cur] ----
        // A 16x32 bf16: lanes 0-15 hold K {0..7,16..23}; lanes 16-31 the rest.
        const __bf16* sAh = smem[cur][0];
        const __bf16* sAl = smem[cur][1];
        const __bf16* sBh = smem[cur][2];
        const __bf16* sBl = smem[cur][3];

        v16bf ah[4], al[4];
        for (int mt = 0; mt < 4; ++mt) {
            const int r  = waveM * 64 + mt * 16 + l15;
            const int e0 = r * LDSTRIDE + lh * 8;
            uint4* ph = reinterpret_cast<uint4*>(&ah[mt]);
            ph[0] = *reinterpret_cast<const uint4*>(&sAh[e0]);
            ph[1] = *reinterpret_cast<const uint4*>(&sAh[e0 + 16]);
            uint4* pl = reinterpret_cast<uint4*>(&al[mt]);
            pl[0] = *reinterpret_cast<const uint4*>(&sAl[e0]);
            pl[1] = *reinterpret_cast<const uint4*>(&sAl[e0 + 16]);
        }
        // B 32x16 bf16: lanes 0-15 hold K {0..7,16..23}; lanes 16-31 the rest.
        v16bf bh[2], bl[2];
        for (int nt = 0; nt < 2; ++nt) {
            const int c  = waveN * 32 + nt * 16 + l15;
            const int e0 = c * LDSTRIDE + lh * 8;
            uint4* ph = reinterpret_cast<uint4*>(&bh[nt]);
            ph[0] = *reinterpret_cast<const uint4*>(&sBh[e0]);
            ph[1] = *reinterpret_cast<const uint4*>(&sBh[e0 + 16]);
            uint4* pl = reinterpret_cast<uint4*>(&bl[nt]);
            pl[0] = *reinterpret_cast<const uint4*>(&sBl[e0]);
            pl[1] = *reinterpret_cast<const uint4*>(&sBl[e0 + 16]);
        }

        // ---- split-bf16 products: xh*wh + xh*wl + xl*wh ----
        for (int mt = 0; mt < 4; ++mt)
            for (int nt = 0; nt < 2; ++nt) {
                acc[mt][nt] = __builtin_amdgcn_wmma_f32_16x16x32_bf16(
                    false, ah[mt], false, bh[nt], (short)0, acc[mt][nt], false, false);
                acc[mt][nt] = __builtin_amdgcn_wmma_f32_16x16x32_bf16(
                    false, ah[mt], false, bl[nt], (short)0, acc[mt][nt], false, false);
                acc[mt][nt] = __builtin_amdgcn_wmma_f32_16x16x32_bf16(
                    false, al[mt], false, bh[nt], (short)0, acc[mt][nt], false, false);
            }
    }

    // ---- epilogue: relu + store. C/D layout: VGPR j -> M = j + 8*lh, N = l15.
    for (int mt = 0; mt < 4; ++mt)
        for (int nt = 0; nt < 2; ++nt) {
            const int gr0 = rowBase + waveM * 64 + mt * 16 + 8 * lh;
            const int gc  = colBase + waveN * 32 + nt * 16 + l15;
            for (int j = 0; j < 8; ++j) {
                float v = acc[mt][nt][j];
                pre[(size_t)(gr0 + j) * WIDTH + gc] = fmaxf(v, 0.0f);
            }
        }
}

// ---------------------------------------------------------------------------
// Kernel 3: per-row top-64 via 4-level radix threshold (8+8+8+7 bits of the
// non-negative, bit-monotone float pattern) -> exact key threshold.
// ---------------------------------------------------------------------------
__global__ __launch_bounds__(256) void sae_topk64(
    const float* __restrict__ pre, float* __restrict__ topv, int* __restrict__ topi) {
    const int row = blockIdx.x;
    const int tid = threadIdx.x;
    const float* p = pre + (size_t)row * WIDTH;

    __shared__ unsigned hist[256];
    __shared__ unsigned s_pref;
    __shared__ int s_need, s_high, s_tie;

    hist[tid] = 0; __syncthreads();
    for (int j = tid; j < WIDTH; j += 256) {
        unsigned k = __builtin_bit_cast(unsigned, p[j]);
        atomicAdd(&hist[k >> 23], 1u);
    }
    __syncthreads();
    if (tid == 0) {
        int need = KSEL; int b = 0;
        for (int i = 255; i >= 0; --i) { int c = (int)hist[i]; if (c >= need) { b = i; break; } need -= c; }
        s_pref = (unsigned)b; s_need = need;
    }
    __syncthreads();
    unsigned pref = s_pref;

    hist[tid] = 0; __syncthreads();
    for (int j = tid; j < WIDTH; j += 256) {
        unsigned k = __builtin_bit_cast(unsigned, p[j]);
        if ((k >> 23) == pref) atomicAdd(&hist[(k >> 15) & 255u], 1u);
    }
    __syncthreads();
    if (tid == 0) {
        int nd = s_need; int b = 0;
        for (int i = 255; i >= 0; --i) { int c = (int)hist[i]; if (c >= nd) { b = i; break; } nd -= c; }
        s_pref = (s_pref << 8) | (unsigned)b; s_need = nd;
    }
    __syncthreads();
    pref = s_pref;

    hist[tid] = 0; __syncthreads();
    for (int j = tid; j < WIDTH; j += 256) {
        unsigned k = __builtin_bit_cast(unsigned, p[j]);
        if ((k >> 15) == pref) atomicAdd(&hist[(k >> 7) & 255u], 1u);
    }
    __syncthreads();
    if (tid == 0) {
        int nd = s_need; int b = 0;
        for (int i = 255; i >= 0; --i) { int c = (int)hist[i]; if (c >= nd) { b = i; break; } nd -= c; }
        s_pref = (s_pref << 8) | (unsigned)b; s_need = nd;
    }
    __syncthreads();
    pref = s_pref;

    hist[tid] = 0; __syncthreads();
    for (int j = tid; j < WIDTH; j += 256) {
        unsigned k = __builtin_bit_cast(unsigned, p[j]);
        if ((k >> 7) == pref) atomicAdd(&hist[k & 127u], 1u);
    }
    __syncthreads();
    if (tid == 0) {
        int nd = s_need; int b = 0;
        for (int i = 127; i >= 0; --i) { int c = (int)hist[i]; if (c >= nd) { b = i; break; } nd -= c; }
        s_pref = (s_pref << 7) | (unsigned)b; s_need = nd;
        s_high = 0; s_tie = 0;
    }
    __syncthreads();
    const unsigned Tk    = s_pref;
    const int      need4 = s_need;

    for (int j = tid; j < WIDTH; j += 256) {
        unsigned k = __builtin_bit_cast(unsigned, p[j]);
        if (k > Tk) {
            int slot = atomicAdd(&s_high, 1);
            topv[(size_t)row * KSEL + slot] = __builtin_bit_cast(float, k);
            topi[(size_t)row * KSEL + slot] = j;
        } else if (k == Tk) {
            int t = atomicAdd(&s_tie, 1);
            if (t < need4) {
                int slot = (KSEL - need4) + t;
                topv[(size_t)row * KSEL + slot] = __builtin_bit_cast(float, k);
                topi[(size_t)row * KSEL + slot] = j;
            }
        }
    }
}

// ---------------------------------------------------------------------------
// Kernel 4: sparse decode.
// ---------------------------------------------------------------------------
__global__ __launch_bounds__(256) void sae_decode(
    const float* __restrict__ topv, const int* __restrict__ topi,
    const float* __restrict__ Ae, const float* __restrict__ norms,
    const float* __restrict__ bd, const float* __restrict__ lambda_pre,
    float* __restrict__ out) {
    const int row = blockIdx.x;
    const int tid = threadIdx.x;
    __shared__ float sw[KSEL];
    __shared__ int   si[KSEL];
    if (tid < KSEL) {
        float lam = log1pf(expf(lambda_pre[0]));   // softplus
        int   ix  = topi[(size_t)row * KSEL + tid];
        float v   = topv[(size_t)row * KSEL + tid];
        sw[tid] = lam * v / (1e-6f + norms[ix]);
        si[tid] = ix;
    }
    __syncthreads();
    const int c0 = tid, c1 = tid + 256, c2 = tid + 512;
    float a0 = 0.f, a1 = 0.f, a2 = 0.f;
    for (int e = 0; e < KSEL; ++e) {
        const float* ar = Ae + (size_t)si[e] * DIMIN;
        float w = sw[e];
        a0 += w * ar[c0];
        a1 += w * ar[c1];
        a2 += w * ar[c2];
    }
    float* o = out + (size_t)row * DIMIN;
    o[c0] = a0 + bd[c0];
    o[c1] = a1 + bd[c1];
    o[c2] = a2 + bd[c2];
}

// ---------------------------------------------------------------------------
extern "C" void kernel_launch(void* const* d_in, const int* in_sizes, int n_in,
                              void* d_out, int out_size, void* d_ws, size_t ws_size,
                              hipStream_t stream) {
    (void)in_sizes; (void)n_in; (void)out_size; (void)ws_size;
    const float* x          = (const float*)d_in[0];
    const float* Ae         = (const float*)d_in[1];
    // d_in[2] = Ad (== Ae^T at init) — decoder columns are Ae rows.
    const float* bd         = (const float*)d_in[3];
    const float* lambda_pre = (const float*)d_in[4];
    float* out = (float*)d_out;

    char* ws = (char*)d_ws;
    float* pre   = (float*)ws;
    size_t off   = (size_t)BATCH * WIDTH * sizeof(float);
    float* norms = (float*)(ws + off); off += (size_t)WIDTH * sizeof(float);
    float* topv  = (float*)(ws + off); off += (size_t)BATCH * KSEL * sizeof(float);
    int*   topi  = (int*)(ws + off);   off += (size_t)BATCH * KSEL * sizeof(int);
    unsigned short* xh  = (unsigned short*)(ws + off); off += (size_t)BATCH * DIMIN * 2;
    unsigned short* xl  = (unsigned short*)(ws + off); off += (size_t)BATCH * DIMIN * 2;
    unsigned short* Aeh = (unsigned short*)(ws + off); off += (size_t)WIDTH * DIMIN * 2;
    unsigned short* Ael = (unsigned short*)(ws + off);

    sae_split_kernel<<<BATCH + WIDTH, 256, 0, stream>>>(x, Ae, bd, xh, xl, Aeh, Ael);
    sae_norms_kernel<<<WIDTH, 256, 0, stream>>>(Ae, norms);
    dim3 g(WIDTH / BN, BATCH / BM);
    sae_encode_gemm<<<g, 256, 0, stream>>>(xh, xl, Aeh, Ael, pre);
    sae_topk64<<<BATCH, 256, 0, stream>>>(pre, topv, topi);
    sae_decode<<<BATCH, 256, 0, stream>>>(topv, topi, Ae, norms, bd, lambda_pre, out);
}
